// NewAttention_6682969112769
// MI455X (gfx1250) — compile-verified, matrix operand order checked by reading, not run on hardware
//
#include <hip/hip_runtime.h>
#include <math.h>

// ---------------------------------------------------------------------------
// MI455X (gfx1250, wave32) implementation.
// ~15 GFLOP total, ~100MB working set (fits global L2) -> structure everything
// as v_wmma_f32_16x16x32_f16 GEMMs with 16B-contiguous fragment loads.
// ---------------------------------------------------------------------------

typedef _Float16 f16;
typedef _Float16 half8  __attribute__((ext_vector_type(8)));
typedef _Float16 half16 __attribute__((ext_vector_type(16)));
typedef float    f32x8  __attribute__((ext_vector_type(8)));

#define NB    8
#define CIN   256
#define COUT  128
#define SDIM  64
#define NTOT  (NB * SDIM * SDIM)   // 32768 spatial*batch columns

// ---- WMMA fragment loader -------------------------------------------------
// A (16x32 f16) lane layout: lane l, half h=l>>4, row m=l&15.
//   elements 0..7  <-> K = 8h + e
//   elements 8..15 <-> K = 16 + 8h + (e-8)
// B operand uses the symmetric striping with lane <-> column, loading from a
// row of the transposed (N-major / K-contiguous) array. Both chunks are
// 16-byte aligned contiguous loads -> global_load_b128.
__device__ __forceinline__ half16 frag_ld(const f16* __restrict__ row, int k0, int hf) {
  half8 lo = *(const half8*)(row + k0 + 8 * hf);
  half8 hi = *(const half8*)(row + k0 + 16 + 8 * hf);
  return __builtin_shufflevector(lo, hi, 0,1,2,3,4,5,6,7,8,9,10,11,12,13,14,15);
}

__device__ __forceinline__ f32x8 wmma_f16(half16 a, half16 b, f32x8 c) {
  return __builtin_amdgcn_wmma_f32_16x16x32_f16(false, a, false, b, (short)0, c, false, false);
}

// ---- elementwise helpers --------------------------------------------------
__global__ void k_cast_f16(const float* __restrict__ src, f16* __restrict__ dst, int n) {
  int i = blockIdx.x * blockDim.x + threadIdx.x;
  if (i < n) dst[i] = (f16)src[i];
}

// xh[((b*64+xx)*64+yy)*256 + ci] = feat[b][ci][yy][xx]  (transpose + channels-last)
__global__ void k_xh(const float* __restrict__ feat, f16* __restrict__ xh) {
  long idx = (long)blockIdx.x * blockDim.x + threadIdx.x;   // = ((b*256+ci)*64+yy)*64+xx
  int xx = idx & 63; long t = idx >> 6;
  int yy = t & 63;   t >>= 6;
  int ci = t & 255;  int b = (int)(t >> 8);
  long n = ((long)b * 64 + xx) * 64 + yy;
  xh[n * CIN + ci] = (f16)feat[idx];
}

__global__ void k_transpose64(const f16* __restrict__ O, f16* __restrict__ OT) {
  int idx = blockIdx.x * blockDim.x + threadIdx.x;          // (b,i,j)
  int j = idx & 63, i = (idx >> 6) & 63, b = idx >> 12;
  OT[b * 4096 + j * 64 + i] = O[idx];
}

// subtract per-(b,spatial) channel mean, in place (planar f16, m-stride 4096)
__global__ void k_meansub(f16* __restrict__ p) {
  int idx = blockIdx.x * blockDim.x + threadIdx.x;          // (b, s)
  int s = idx & 4095, b = idx >> 12;
  f16* q = p + (long)b * COUT * 4096 + s;
  float sum = 0.f;
  for (int m = 0; m < COUT; ++m) sum += (float)q[(long)m * 4096];
  float mean = sum * (1.0f / COUT);
  for (int m = 0; m < COUT; ++m) {
    long o = (long)m * 4096;
    q[o] = (f16)((float)q[o] - mean);
  }
}

// column softmax of cov[b][64][64] over rows, write transposed f16 (row j = softmaxed column j)
__global__ void k_softmax_colT(const float* __restrict__ cov, f16* __restrict__ outT) {
  int idx = blockIdx.x * blockDim.x + threadIdx.x;          // (b, j)
  int j = idx & 63, b = idx >> 6;
  const float* p = cov + b * 4096 + j;
  float mx = -3.4e38f;
  for (int i = 0; i < 64; ++i) { float v = p[i * 64]; mx = v > mx ? v : mx; }
  float s = 0.f;
  for (int i = 0; i < 64; ++i) s += expf(p[i * 64] - mx);
  float inv = 1.0f / s;
  for (int i = 0; i < 64; ++i)
    outT[b * 4096 + j * 64 + i] = (f16)(expf(p[i * 64] - mx) * inv);
}

// row softmax of cov[b][64][64], write f16 row-major
__global__ void k_softmax_row(const float* __restrict__ cov, f16* __restrict__ out) {
  int idx = blockIdx.x * blockDim.x + threadIdx.x;          // (b, i)
  int i = idx & 63, b = idx >> 6;
  const float* p = cov + b * 4096 + i * 64;
  float mx = -3.4e38f;
  for (int j = 0; j < 64; ++j) { float v = p[j]; mx = v > mx ? v : mx; }
  float s = 0.f;
  for (int j = 0; j < 64; ++j) s += expf(p[j] - mx);
  float inv = 1.0f / s;
  for (int j = 0; j < 64; ++j)
    out[b * 4096 + i * 64 + j] = (f16)(expf(p[j] - mx) * inv);
}

// ---- direct 3x3 conv + BN + ReLU (small channel counts; VALU) -------------
template<bool F32IN>
__global__ void k_conv3(const void* __restrict__ inv, const float* __restrict__ w,
                        const float* __restrict__ sc, const float* __restrict__ bi,
                        f16* __restrict__ out, int Ci, int Co, int Hin, int stride) {
  int Hout = (Hin - 1) / stride + 1;
  long total = (long)NB * Co * Hout * Hout;
  long idx = (long)blockIdx.x * blockDim.x + threadIdx.x;
  if (idx >= total) return;
  int ow = (int)(idx % Hout); long t = idx / Hout;
  int oh = (int)(t % Hout);   t /= Hout;
  int co = (int)(t % Co);     int b = (int)(t / Co);
  const float* wv = w + (long)co * Ci * 9;
  float acc = 0.f;
  for (int ci = 0; ci < Ci; ++ci) {
    long ibase = (((long)b * Ci + ci) * Hin) * Hin;
    for (int kh = 0; kh < 3; ++kh) {
      int ih = oh * stride + kh - 1;
      if ((unsigned)ih >= (unsigned)Hin) continue;
      for (int kw = 0; kw < 3; ++kw) {
        int iw = ow * stride + kw - 1;
        if ((unsigned)iw >= (unsigned)Hin) continue;
        long off = ibase + (long)ih * Hin + iw;
        float v = F32IN ? ((const float*)inv)[off] : (float)((const f16*)inv)[off];
        acc += v * wv[ci * 9 + kh * 3 + kw];
      }
    }
  }
  acc = acc * sc[co] + bi[co];
  out[idx] = (f16)(acc > 0.f ? acc : 0.f);
}

// ---- branch 1x1-conv GEMM, channels-last output ([n][128] f16) ------------
// A: W [128][K] row-major f16.  X: [N][K] f16 (K-contiguous).  One wave/tile.
template<int K>
__global__ void k_gemm_bnrelu_cl(const f16* __restrict__ A, const f16* __restrict__ X,
                                 const float* __restrict__ sc, const float* __restrict__ bi,
                                 f16* __restrict__ out) {
  int wid = threadIdx.x >> 5, lane = threadIdx.x & 31;
  int lm = lane & 15, hf = lane >> 4;
  int w = blockIdx.x * 8 + wid;
  int nt = w >> 3, mt = w & 7;                 // 2048 n-tiles x 8 m-tiles
  const f16* arow = A + (long)(mt * 16 + lm) * K;
  const f16* brow = X + (long)(nt * 16 + lm) * K;
  f32x8 acc = {0.f,0.f,0.f,0.f,0.f,0.f,0.f,0.f};
  for (int k0 = 0; k0 < K; k0 += 32)
    acc = wmma_f16(frag_ld(arow, k0, hf), frag_ld(brow, k0, hf), acc);
  int n = nt * 16 + lm;
  int mbase = mt * 16 + 8 * hf;                // C frag: row m = mt*16 + r + 8*hf
  half8 o;
#pragma unroll
  for (int r = 0; r < 8; ++r) {
    float v = acc[r] * sc[mbase + r] + bi[mbase + r];
    o[r] = (f16)(v > 0.f ? v : 0.f);
  }
  *(half8*)(out + (long)n * COUT + mbase) = o;  // 16B packed store
}

// ---- branch conv2 GEMM, per-channel 64x64 matrix output -------------------
// out[(b*128+m)*4096 + (transpose ? y*64+x : x*64+y)], n = ((b*64+x)*64+y)
template<int K>
__global__ void k_gemm_bnrelu_planar(const f16* __restrict__ A, const f16* __restrict__ X,
                                     const float* __restrict__ sc, const float* __restrict__ bi,
                                     f16* __restrict__ out, int transposeOut) {
  int wid = threadIdx.x >> 5, lane = threadIdx.x & 31;
  int lm = lane & 15, hf = lane >> 4;
  int w = blockIdx.x * 8 + wid;
  int nt = w >> 3, mt = w & 7;
  const f16* arow = A + (long)(mt * 16 + lm) * K;
  const f16* brow = X + (long)(nt * 16 + lm) * K;
  f32x8 acc = {0.f,0.f,0.f,0.f,0.f,0.f,0.f,0.f};
  for (int k0 = 0; k0 < K; k0 += 32)
    acc = wmma_f16(frag_ld(arow, k0, hf), frag_ld(brow, k0, hf), acc);
  int n = nt * 16 + lm;
  int b = n >> 12, x = (n >> 6) & 63, y = n & 63;
  long sp = transposeOut ? (y * 64 + x) : (x * 64 + y);
#pragma unroll
  for (int r = 0; r < 8; ++r) {
    int m = mt * 16 + 8 * hf + r;
    float v = acc[r] * sc[m] + bi[m];
    out[((long)(b * COUT + m)) * 4096 + sp] = (f16)(v > 0.f ? v : 0.f);
  }
}

// ---- generic per-(b,m) 64x64x64 WMMA GEMM ---------------------------------
// A[i][k] at A0 + b*aSb + m*aSm (row-major, ld 64); BT[j][k] likewise.
// OM: 0 = f16 row-major, 1 = f16 transposed (16B packed), 2 = f32 row-major.
template<int OM>
__global__ void k_gemm64(const f16* __restrict__ A0, long aSb, long aSm,
                         const f16* __restrict__ B0, long bSb, long bSm,
                         void* __restrict__ O0) {
  int wid = threadIdx.x >> 5, lane = threadIdx.x & 31;
  int lm = lane & 15, hf = lane >> 4;
  int w = blockIdx.x * 8 + wid;                // 16384 waves
  int chan = w >> 4, t = w & 15, ti = t >> 2, tj = t & 3;
  int b = chan >> 7, m = chan & 127;
  const f16* ar = A0 + (long)b * aSb + (long)m * aSm + (long)(ti * 16 + lm) * 64;
  const f16* br = B0 + (long)b * bSb + (long)m * bSm + (long)(tj * 16 + lm) * 64;
  f32x8 acc = {0.f,0.f,0.f,0.f,0.f,0.f,0.f,0.f};
  acc = wmma_f16(frag_ld(ar, 0, hf),  frag_ld(br, 0, hf),  acc);
  acc = wmma_f16(frag_ld(ar, 32, hf), frag_ld(br, 32, hf), acc);
  if (OM == 1) {
    f16* O = (f16*)O0 + (long)chan * 4096;
    half8 o;
#pragma unroll
    for (int r = 0; r < 8; ++r) o[r] = (f16)acc[r];
    *(half8*)(O + (tj * 16 + lm) * 64 + ti * 16 + 8 * hf) = o;
  } else if (OM == 0) {
    f16* O = (f16*)O0 + (long)chan * 4096;
#pragma unroll
    for (int r = 0; r < 8; ++r)
      O[(ti * 16 + 8 * hf + r) * 64 + tj * 16 + lm] = (f16)acc[r];
  } else {
    float* O = (float*)O0 + (long)chan * 4096;
#pragma unroll
    for (int r = 0; r < 8; ++r)
      O[(ti * 16 + 8 * hf + r) * 64 + tj * 16 + lm] = acc[r];
  }
}

// ---- channel-mean covariance: out[b] = (1/128) sum_m X[b,m] * X[b,m]^T-style
// Both fragments read rows of the same per-channel matrix (layout pre-arranged
// so this yields SR^T·SR for subtra_r and SC·SC^T for subtra_c).
__global__ void k_cov(const f16* __restrict__ X0, float* __restrict__ out) {
  int wid = threadIdx.x >> 5, lane = threadIdx.x & 31;
  int lm = lane & 15, hf = lane >> 4;
  int w = blockIdx.x * 8 + wid;                // 128 waves
  int b = w >> 4, t = w & 15, ti = t >> 2, tj = t & 3;
  f32x8 acc = {0.f,0.f,0.f,0.f,0.f,0.f,0.f,0.f};
  for (int m = 0; m < COUT; ++m) {
    const f16* Xm = X0 + ((long)(b * COUT + m)) * 4096;
    const f16* ar = Xm + (ti * 16 + lm) * 64;
    const f16* br = Xm + (tj * 16 + lm) * 64;
    acc = wmma_f16(frag_ld(ar, 0, hf),  frag_ld(br, 0, hf),  acc);
    acc = wmma_f16(frag_ld(ar, 32, hf), frag_ld(br, 32, hf), acc);
  }
  float* O = out + b * 4096;
#pragma unroll
  for (int r = 0; r < 8; ++r)
    O[(ti * 16 + 8 * hf + r) * 64 + tj * 16 + lm] = acc[r] * (1.0f / COUT);
}

// ---------------------------------------------------------------------------
extern "C" void kernel_launch(void* const* d_in, const int* in_sizes, int n_in,
                              void* d_out, int out_size, void* d_ws, size_t ws_size,
                              hipStream_t stream) {
  const float* feat      = (const float*)d_in[0];
  const float* otherfeat = (const float*)d_in[1];
  const float* w1_rbc    = (const float*)d_in[2];
  const float* s1_rbc    = (const float*)d_in[3];
  const float* b1_rbc    = (const float*)d_in[4];
  const float* w2_rbc    = (const float*)d_in[5];
  const float* s2_rbc    = (const float*)d_in[6];
  const float* b2_rbc    = (const float*)d_in[7];
  const float* ow[6]; const float* os[6]; const float* ob[6];
  for (int i = 0; i < 6; ++i) { ow[i] = (const float*)d_in[8 + 3*i];
                                os[i] = (const float*)d_in[9 + 3*i];
                                ob[i] = (const float*)d_in[10 + 3*i]; }

  char* ws = (char*)d_ws;
  // phase-aliased pool (o-stack intermediates, then xh/hbuf, then srt/sc/tmp)
  f16* T1   = (f16*)(ws + 0);           // [8,32,256,256] 33.5MB
  f16* T2   = (f16*)(ws + 33554432);    // [8,1,256,256]  1.0MB
  f16* T3   = (f16*)(ws + 34603008);    // [8,32,128,128] 8.4MB
  f16* T4   = (f16*)(ws + 42991616);    // [8,1,128,128]  0.26MB
  f16* T5   = (f16*)(ws + 43253760);    // [8,32,64,64]   2.1MB
  f16* XH   = (f16*)(ws + 0);           // [32768,256]    16.8MB  (after o-stack)
  f16* HBUF = (f16*)(ws + 16777216);    // [32768,128]    8.4MB
  f16* SRT  = (f16*)(ws + 0);           // [8,128,64,64]  8.4MB   (after branches)
  f16* SC   = (f16*)(ws + 8388608);     // [8,128,64,64]  8.4MB
  f16* TMP  = (f16*)(ws + 16777216);    // [8,128,64,64]  8.4MB
  size_t P = 45350912;                  // persistent region
  f16*   W1H   = (f16*)(ws + P);             // 3*128*256
  f16*   W2H   = (f16*)(ws + P + 196608);    // 3*128*128
  f16*   RBUF  = (f16*)(ws + P + 294912);    // rT  [b][m][y][x]
  f16*   BBUF  = (f16*)(ws + P + 8683520);   // bT  [b][m][y][x]
  f16*   CBUF  = (f16*)(ws + P + 17072128);  // c   [b][m][x][y]
  f16*   O16   = (f16*)(ws + P + 25460736);  // o   [b][64][64]
  f16*   OT16  = (f16*)(ws + P + 25526272);  // o^T
  float* COVR  = (float*)(ws + P + 25591808);
  float* COVC  = (float*)(ws + P + 25722880);
  f16*   HSRT  = (f16*)(ws + P + 25853952);  // hsr transposed
  f16*   HSC16 = (f16*)(ws + P + 25886720);  // hsc row-major

  // 1) o conv stack: 512->256->256->128->128->64->64
  k_conv3<true ><<<65536, 256, 0, stream>>>(otherfeat, ow[0], os[0], ob[0], T1, 3, 32, 512, 2);
  k_conv3<false><<< 2048, 256, 0, stream>>>(T1, ow[1], os[1], ob[1], T2, 32, 1, 256, 1);
  k_conv3<false><<<16384, 256, 0, stream>>>(T2, ow[2], os[2], ob[2], T3, 1, 32, 256, 2);
  k_conv3<false><<<  512, 256, 0, stream>>>(T3, ow[3], os[3], ob[3], T4, 32, 1, 128, 1);
  k_conv3<false><<< 4096, 256, 0, stream>>>(T4, ow[4], os[4], ob[4], T5, 1, 32, 128, 2);
  k_conv3<false><<<  128, 256, 0, stream>>>(T5, ow[5], os[5], ob[5], O16, 32, 1, 64, 1);
  k_transpose64<<<128, 256, 0, stream>>>(O16, OT16);

  // 2) f16 conversions
  k_cast_f16<<<384, 256, 0, stream>>>(w1_rbc, W1H, 3 * COUT * CIN);
  k_cast_f16<<<192, 256, 0, stream>>>(w2_rbc, W2H, 3 * COUT * COUT);
  k_xh<<<32768, 256, 0, stream>>>(feat, XH);

  // 3) branches r(0,T), b(1,T), c(2,normal)
  f16* dst[3] = {RBUF, BBUF, CBUF};
  int  trs[3] = {1, 1, 0};
  for (int i = 0; i < 3; ++i) {
    k_gemm_bnrelu_cl<CIN><<<2048, 256, 0, stream>>>(
        W1H + (long)i * COUT * CIN, XH, s1_rbc + i * COUT, b1_rbc + i * COUT, HBUF);
    k_gemm_bnrelu_planar<COUT><<<2048, 256, 0, stream>>>(
        W2H + (long)i * COUT * COUT, HBUF, s2_rbc + i * COUT, b2_rbc + i * COUT, dst[i], trs[i]);
  }

  // 4) subtract channel means (r and c only)
  k_meansub<<<128, 256, 0, stream>>>(RBUF);
  k_meansub<<<128, 256, 0, stream>>>(CBUF);

  // 5) subtra_r = o @ rm  (store transposed -> both cov fragments contiguous)
  k_gemm64<1><<<2048, 256, 0, stream>>>(O16, 4096, 0, RBUF, (long)COUT * 4096, 4096, SRT);
  k_cov<<<16, 256, 0, stream>>>(SRT, COVR);
  k_softmax_colT<<<2, 256, 0, stream>>>(COVR, HSRT);

  // 6) subtra_c = cm @ o
  k_gemm64<0><<<2048, 256, 0, stream>>>(CBUF, (long)COUT * 4096, 4096, OT16, 4096, 0, SC);
  k_cov<<<16, 256, 0, stream>>>(SC, COVC);
  k_softmax_row<<<2, 256, 0, stream>>>(COVC, HSC16);

  // 7) out = (hsc @ b) @ hsr
  k_gemm64<0><<<2048, 256, 0, stream>>>(HSC16, 4096, 0, BBUF, (long)COUT * 4096, 4096, TMP);
  k_gemm64<2><<<2048, 256, 0, stream>>>(TMP, (long)COUT * 4096, 4096, HSRT, 4096, 0, d_out);
}
